// AdaptiveDecision_50646254354922
// MI455X (gfx1250) — compile-verified, matrix operand order checked by reading, not run on hardware
//
#include <hip/hip_runtime.h>

// ---------------------------------------------------------------------------
// Fused AdaptiveDecision block for MI455X (gfx1250, wave32, WMMA bf16)
//   B=32768, C=1024, CH=256, rank=16 (padded to 32)
// ---------------------------------------------------------------------------

#define CDIM   1024
#define CHDIM  256
#define ROWS   64            // rows per workgroup
#define NWAVE  8             // 256 threads / 32

typedef __attribute__((ext_vector_type(16))) __bf16 v16bf;
typedef __attribute__((ext_vector_type(8)))  float  v8f;

// ---- bf16 <-> f32 (manual RNE, storage as ushort) -------------------------
__device__ __forceinline__ unsigned short f2bf(float f) {
  unsigned u = __float_as_uint(f);
  u += 0x7FFFu + ((u >> 16) & 1u);
  return (unsigned short)(u >> 16);
}
__device__ __forceinline__ float bf2f(unsigned short h) {
  return __uint_as_float((unsigned)h << 16);
}

// Position (lane*16 + halfIdx) inside a 512-ushort WMMA fragment tile for
// element (k32 = K index within 32-wide K-tile, m16 = row/col within 16).
// Derived from the CDNA5 ISA 16-bit A/B VGPR layout tables.
__device__ __forceinline__ int frag_pos(int k32, int m16) {
  int p  =  k32       & 1;
  int vl = (k32 >> 1) & 3;
  int hi = (k32 >> 3) & 1;
  int vh = (k32 >> 4) & 1;
  int v  = vh * 4 + vl;
  return (hi * 16 + m16) * 16 + (2 * v + p);
}

__device__ __forceinline__ v16bf load_frag(const unsigned short* base, int tile, int lane) {
  return *reinterpret_cast<const v16bf*>(base + (size_t)tile * 512 + lane * 16);
}

__device__ __forceinline__ float gelu_f(float x) {
  float x3 = x * x * x;
  return 0.5f * x * (1.0f + tanhf(0.7978845608028654f * (x + 0.044715f * x3)));
}
__device__ __forceinline__ float sigmoid_f(float x) {
  return 1.0f / (1.0f + __expf(-x));
}

// ---------------------------------------------------------------------------
// Prep: repack fp32 [K,N] (row-major, K = "in", N = "out") into bf16 WMMA
// fragment tiles of 32x16 (1 KB each), tile index = kt*(N/16)+nt.
// Rows k >= Ksrc are zero-filled (LoRA K padding 16 -> 32).
// ---------------------------------------------------------------------------
__global__ void pack_w_kernel(const float* __restrict__ src,
                              unsigned short* __restrict__ dst,
                              int K, int N, int Ksrc) {
  int idx = blockIdx.x * blockDim.x + threadIdx.x;
  if (idx >= K * N) return;
  int k = idx / N, n = idx % N;
  float v = (k < Ksrc) ? src[(size_t)k * N + n] : 0.0f;
  int tile = (k >> 5) * (N >> 4) + (n >> 4);
  dst[(size_t)tile * 512 + frag_pos(k & 31, n & 15)] = f2bf(v);
}

// ---------------------------------------------------------------------------
// Generic fragment-domain GEMM stage:
//   out[MTx16, NTx16] = epilogue( in[MTx16, KTx32] @ W[KTx32, NTx16] + bias )
// in/out are LDS fragment buffers (A-layout), W is packed global bf16.
// Output scattered back into A-layout fragments with KTout K-tiles per row.
// ---------------------------------------------------------------------------
template <bool GELU>
__device__ __forceinline__ void gemm_stage(const unsigned short* __restrict__ lin,
                                           unsigned short* __restrict__ lout,
                                           const unsigned short* __restrict__ W,
                                           const float* __restrict__ bias,
                                           int MT, int KT, int NT, int KTout,
                                           int wave, int lane) {
  const int total = MT * NT;
  for (int t = wave; t < total; t += NWAVE) {
    int mt = t / NT, nt = t % NT;
    v8f acc = {};
    for (int kt = 0; kt < KT; ++kt) {
      v16bf a = load_frag(lin, mt * KT + kt, lane);
      v16bf b = load_frag(W,   kt * NT + nt, lane);
      acc = __builtin_amdgcn_wmma_f32_16x16x32_bf16(false, a, false, b,
                                                    (short)0, acc, false, false);
    }
    int n_l = lane & 15, hi = lane >> 4;
    int ng = nt * 16 + n_l;
    float bs = bias ? bias[ng] : 0.0f;
    int kt2 = ng >> 5, k32 = ng & 31;
#pragma unroll
    for (int r = 0; r < 8; ++r) {
      int m_l = r + 8 * hi;
      float v = acc[r] + bs;
      if (GELU) v = gelu_f(v);
      lout[(size_t)(mt * KTout + kt2) * 512 + frag_pos(k32, m_l)] = f2bf(v);
    }
  }
}

// ---------------------------------------------------------------------------
// Fused kernel: LN -> gated down-proj -> dw affine -> FFN -> V,O -> up-proj
//               -> LoRA -> residual mix.  One block = 64 rows.
// Dynamic LDS layout (ushort units):
//   [0      , 65536)  region A : xn frags (4mt x 32kt), reused for up-proj out
//   [65536  , 81920)  region B : CH-wide frags (4mt x 8kt) ping
//   [81920  , 98304)  region C : CH-wide frags pong
//   [98304  , 100352) region L : LoRA rank frags (4mt x 1kt)
// ---------------------------------------------------------------------------
__global__ __launch_bounds__(256) void fused_kernel(
    const float* __restrict__ x,
    const float* __restrict__ ln_g, const float* __restrict__ ln_b,
    const float* __restrict__ bd,   const float* __restrict__ bg,
    const float* __restrict__ dw_w, const float* __restrict__ dw_b,
    const float* __restrict__ b1,   const float* __restrict__ b2,
    const float* __restrict__ bv,   const float* __restrict__ bo,
    const float* __restrict__ bu,
    const unsigned short* __restrict__ Wdp, const unsigned short* __restrict__ Wgp,
    const unsigned short* __restrict__ W1p, const unsigned short* __restrict__ W2p,
    const unsigned short* __restrict__ Wvp, const unsigned short* __restrict__ Wop,
    const unsigned short* __restrict__ Wup, const unsigned short* __restrict__ Wldp,
    const unsigned short* __restrict__ Wlup,
    float* __restrict__ out) {
  extern __shared__ unsigned short smem[];
  unsigned short* smA = smem;               // 65536
  unsigned short* smB = smem + 65536;       // 16384
  unsigned short* smC = smem + 81920;       // 16384
  unsigned short* smL = smem + 98304;       //  2048

  const int tid = threadIdx.x;
  const int wave = tid >> 5, lane = tid & 31;
  const int row0 = blockIdx.x * ROWS;

  // ---- Stage 0: LayerNorm, scatter bf16 into A-fragment layout ----
  for (int rr = 0; rr < 8; ++rr) {
    int row = wave * 8 + rr;
    const float4* xr = reinterpret_cast<const float4*>(x + (size_t)(row0 + row) * CDIM);
    float4 vals[8];
    float s = 0.f, s2 = 0.f;
#pragma unroll
    for (int i = 0; i < 8; ++i) {
      float4 v = xr[i * 32 + lane];
      vals[i] = v;
      s  += v.x + v.y + v.z + v.w;
      s2 += v.x * v.x + v.y * v.y + v.z * v.z + v.w * v.w;
    }
#pragma unroll
    for (int off = 16; off > 0; off >>= 1) {
      s  += __shfl_xor(s,  off, 32);
      s2 += __shfl_xor(s2, off, 32);
    }
    float mu = s * (1.0f / CDIM);
    float rs = rsqrtf(s2 * (1.0f / CDIM) - mu * mu + 1e-5f);
    int mt = row >> 4, m_l = row & 15;
#pragma unroll
    for (int i = 0; i < 8; ++i) {
      float4 g = reinterpret_cast<const float4*>(ln_g)[i * 32 + lane];
      float4 bb = reinterpret_cast<const float4*>(ln_b)[i * 32 + lane];
      float vv[4] = {vals[i].x, vals[i].y, vals[i].z, vals[i].w};
      float gg[4] = {g.x, g.y, g.z, g.w};
      float bv4[4] = {bb.x, bb.y, bb.z, bb.w};
      int c0 = (i * 32 + lane) * 4;
#pragma unroll
      for (int e = 0; e < 4; ++e) {
        int c = c0 + e;
        float h = (vv[e] - mu) * rs * gg[e] + bv4[e];
        smA[(size_t)(mt * 32 + (c >> 5)) * 512 + frag_pos(c & 31, m_l)] = f2bf(h);
      }
    }
  }
  __syncthreads();

  // ---- Stage 1: gated down-proj + depthwise affine (dual-accumulator) ----
  for (int t = wave; t < 4 * 16; t += NWAVE) {
    int mt = t / 16, nt = t % 16;
    v8f ad = {}, ag = {};
    for (int kt = 0; kt < 32; ++kt) {
      v16bf a  = load_frag(smA, mt * 32 + kt, lane);
      v16bf wd = load_frag(Wdp, kt * 16 + nt, lane);
      ad = __builtin_amdgcn_wmma_f32_16x16x32_bf16(false, a, false, wd, (short)0, ad, false, false);
      v16bf wg = load_frag(Wgp, kt * 16 + nt, lane);
      ag = __builtin_amdgcn_wmma_f32_16x16x32_bf16(false, a, false, wg, (short)0, ag, false, false);
    }
    int n_l = lane & 15, hi = lane >> 4;
    int ng = nt * 16 + n_l;
    float bdv = bd[ng], bgv = bg[ng], dwv = dw_w[ng], dbv = dw_b[ng];
    int kt2 = ng >> 5, k32 = ng & 31;
#pragma unroll
    for (int r = 0; r < 8; ++r) {
      int m_l = r + 8 * hi;
      float h = (ad[r] + bdv) * sigmoid_f(ag[r] + bgv);
      h = h * dwv + dbv;
      smB[(size_t)(mt * 8 + kt2) * 512 + frag_pos(k32, m_l)] = f2bf(h);
    }
  }
  __syncthreads();

  // ---- FFN + attention value/output path (q,k are dead: softmax(1)=1) ----
  gemm_stage<true >(smB, smC, W1p, b1, 4, 8, 16, 8, wave, lane); __syncthreads();
  gemm_stage<false>(smC, smB, W2p, b2, 4, 8, 16, 8, wave, lane); __syncthreads();
  gemm_stage<false>(smB, smC, Wvp, bv, 4, 8, 16, 8, wave, lane); __syncthreads();
  gemm_stage<false>(smC, smB, Wop, bo, 4, 8, 16, 8, wave, lane); __syncthreads();

  // zero LoRA rank frag region (K 16..31 padding), then up-projection into A
  for (int i = tid; i < 2048; i += 256) smL[i] = 0;
  gemm_stage<false>(smB, smA, Wup, bu, 4, 8, 64, 32, wave, lane); __syncthreads();

  // ---- LoRA down: l = h_up @ Wld (N=16, one tile column) ----
  gemm_stage<false>(smA, smL, Wldp, nullptr, 4, 32, 1, 1, wave, lane); __syncthreads();

  // ---- LoRA up + residual mix, write fp32 output ----
  for (int t = wave; t < 4 * 64; t += NWAVE) {
    int mt = t >> 6, nt = t & 63;
    v16bf a = load_frag(smL, mt, lane);
    v16bf b = load_frag(Wlup, nt, lane);
    v8f acc = {};
    acc = __builtin_amdgcn_wmma_f32_16x16x32_bf16(false, a, false, b, (short)0, acc, false, false);
    int n_l = lane & 15, hi = lane >> 4;
    int ng = nt * 16 + n_l;
    int kt2 = ng >> 5, k32 = ng & 31;
#pragma unroll
    for (int r = 0; r < 8; ++r) {
      int m_l = r + 8 * hi;
      int rowg = row0 + mt * 16 + m_l;
      float hup = bf2f(smA[(size_t)(mt * 32 + kt2) * 512 + frag_pos(k32, m_l)]);
      float xv = x[(size_t)rowg * CDIM + ng];
      out[(size_t)rowg * CDIM + ng] = 0.5f * (hup + acc[r]) + 0.5f * xv;
    }
  }
}

// ---------------------------------------------------------------------------
// Workspace layout (ushort units)
// ---------------------------------------------------------------------------
#define OFF_WD   0u
#define OFF_WG   262144u
#define OFF_W1   524288u
#define OFF_W2   589824u
#define OFF_WV   655360u
#define OFF_WO   720896u
#define OFF_WU   786432u
#define OFF_WLD  1048576u
#define OFF_WLU  1064960u
// total 1097728 ushort = ~2.1 MB

extern "C" void kernel_launch(void* const* d_in, const int* in_sizes, int n_in,
                              void* d_out, int out_size, void* d_ws, size_t ws_size,
                              hipStream_t stream) {
  (void)n_in; (void)out_size; (void)ws_size;
  const float* x    = (const float*)d_in[0];
  const float* ln_g = (const float*)d_in[1];
  const float* ln_b = (const float*)d_in[2];
  const float* Wd   = (const float*)d_in[3];
  const float* bd   = (const float*)d_in[4];
  const float* Wg   = (const float*)d_in[5];
  const float* bg   = (const float*)d_in[6];
  const float* dw_w = (const float*)d_in[7];
  const float* dw_b = (const float*)d_in[8];
  const float* W1   = (const float*)d_in[9];
  const float* b1   = (const float*)d_in[10];
  const float* W2   = (const float*)d_in[11];
  const float* b2   = (const float*)d_in[12];
  // d_in[13..16] = Wq,bq,Wk,bk : mathematically dead (softmax over 1 key)
  const float* Wv   = (const float*)d_in[17];
  const float* bv   = (const float*)d_in[18];
  const float* Wo   = (const float*)d_in[19];
  const float* bo   = (const float*)d_in[20];
  const float* Wu   = (const float*)d_in[21];
  const float* bu   = (const float*)d_in[22];
  const float* Wld  = (const float*)d_in[23];
  const float* Wlu  = (const float*)d_in[24];

  unsigned short* ws = (unsigned short*)d_ws;
  float* out = (float*)d_out;

  auto pack = [&](const float* src, unsigned int off, int K, int N, int Ksrc) {
    int total = K * N;
    pack_w_kernel<<<(total + 255) / 256, 256, 0, stream>>>(src, ws + off, K, N, Ksrc);
  };
  pack(Wd,  OFF_WD,  1024, 256, 1024);
  pack(Wg,  OFF_WG,  1024, 256, 1024);
  pack(W1,  OFF_W1,  256,  256, 256);
  pack(W2,  OFF_W2,  256,  256, 256);
  pack(Wv,  OFF_WV,  256,  256, 256);
  pack(Wo,  OFF_WO,  256,  256, 256);
  pack(Wu,  OFF_WU,  256, 1024, 256);
  pack(Wld, OFF_WLD, 1024, 16, 1024);
  pack(Wlu, OFF_WLU, 32, 1024, 16);   // K padded 16 -> 32 with zero rows

  const int n_rows = in_sizes[0] / CDIM;      // 32768
  const int grid = n_rows / ROWS;             // 512
  const size_t smem_bytes = 100352u * sizeof(unsigned short);  // 200704 B

  fused_kernel<<<grid, 256, smem_bytes, stream>>>(
      x, ln_g, ln_b, bd, bg, dw_w, dw_b, b1, b2, bv, bo, bu,
      ws + OFF_WD, ws + OFF_WG, ws + OFF_W1, ws + OFF_W2,
      ws + OFF_WV, ws + OFF_WO, ws + OFF_WU, ws + OFF_WLD, ws + OFF_WLU,
      out);
}
